// UNet_13546326851841
// MI455X (gfx1250) — matrix operand order repaired and checked by reading, not compiled, not run on hardware
//
#include <hip/hip_runtime.h>
#include <hip/hip_bf16.h>

// ---------------- types ----------------
typedef __bf16 bf16_t;
typedef __attribute__((ext_vector_type(16))) __bf16 v16bf;
typedef __attribute__((ext_vector_type(8)))  __bf16 v8bf;
typedef __attribute__((ext_vector_type(8)))  float  v8f;
typedef int v4i_t __attribute__((__vector_size__(16)));

#define GAS __attribute__((address_space(1)))
#define LAS __attribute__((address_space(3)))

#if __has_builtin(__builtin_amdgcn_global_load_async_to_lds_b128) && \
    __has_builtin(__builtin_amdgcn_s_wait_asynccnt)
#define HAVE_ASYNC_LDS 1
#else
#define HAVE_ASYNC_LDS 0
#endif

#if HAVE_ASYNC_LDS
__device__ __forceinline__ void async_copy16(const void* g, void* l) {
  __builtin_amdgcn_global_load_async_to_lds_b128((GAS v4i_t*)g, (LAS v4i_t*)l, 0, 0);
}
#endif

// ---------------- config ----------------
#define CC    128   // C_IN == C_OUT
#define HIMG  256
#define WIMG  128
#define BNUM  2
#define RR    8
#define NF    9     // 3x3
#define ZD    128
#define HIDD  128
#define SD    48
#define HO    64
#define NFIN  4608  // 64 * 72

__device__ __forceinline__ v8f v8f_zero() {
  v8f z;
  #pragma unroll
  for (int i = 0; i < 8; ++i) z[i] = 0.0f;
  return z;
}

__device__ __forceinline__ v8bf v8bf_zero() {
  v8bf z;
  #pragma unroll
  for (int i = 0; i < 8; ++i) z[i] = (bf16_t)0.0f;
  return z;
}

__device__ __forceinline__ v16bf cat8(v8bf lo, v8bf hi) {
  return __builtin_shufflevector(lo, hi, 0,1,2,3,4,5,6,7,8,9,10,11,12,13,14,15);
}

// ================= K1: MLP bank =================
// grid 16 blocks (g = ((b*2+lvl)*4 + m)), 128 threads
__global__ void mlp_kernel(const float* __restrict__ seidel,
                           const float* __restrict__ w1a, const float* __restrict__ b1a,
                           const float* __restrict__ w2a, const float* __restrict__ b2a,
                           const float* __restrict__ w1b, const float* __restrict__ b1b,
                           const float* __restrict__ w2b, const float* __restrict__ b2b,
                           float* __restrict__ emb) {
  __shared__ float zs[SD];
  __shared__ float hs[HIDD];
  int g = blockIdx.x;
  int b = g >> 3, lvl = (g >> 2) & 1, m = g & 3;
  const float* W1 = lvl ? w1b : w1a;
  const float* B1 = lvl ? b1b : b1a;
  const float* W2 = lvl ? w2b : w2a;
  const float* B2 = lvl ? b2b : b2a;
  int tid = threadIdx.x;
  if (tid < SD) zs[tid] = seidel[b * SD + tid];
  __syncthreads();
  float a1 = B1[m * HIDD + tid];
  #pragma unroll 4
  for (int s = 0; s < SD; ++s) a1 += zs[s] * W1[(m * SD + s) * HIDD + tid];
  hs[tid] = fmaxf(a1, 0.0f);
  __syncthreads();
  float a2 = B2[m * HIDD + tid];
  #pragma unroll 4
  for (int j = 0; j < HIDD; ++j) a2 += hs[j] * W2[(m * HIDD + j) * HIDD + tid];
  emb[g * HIDD + tid] = fmaxf(a2, 0.0f);
}

// ================= K2: h_in = emb @ hyp_w2 + hb2 (bf16 out) =================
// grid (32, 16), 256 threads
__global__ void hin_kernel(const float* __restrict__ emb,
                           const float* __restrict__ hw2, const float* __restrict__ hb2,
                           bf16_t* __restrict__ hin) {
  __shared__ float es[ZD];
  int g = blockIdx.y;
  int col = blockIdx.x * 256 + threadIdx.x;
  if (threadIdx.x < ZD) es[threadIdx.x] = emb[g * ZD + threadIdx.x];
  __syncthreads();
  float a = hb2[col];
  #pragma unroll 4
  for (int k = 0; k < ZD; ++k) a += es[k] * hw2[k * (HO * ZD) + col];
  hin[g * (HO * ZD) + col] = (bf16_t)a;
}

// ================= K2b: f32 -> bf16 convert =================
__global__ void cvt_kernel(const float* __restrict__ src, bf16_t* __restrict__ dst, int n) {
  int i = blockIdx.x * 256 + threadIdx.x;
  if (i < n) dst[i] = (bf16_t)src[i];
}

// ================= K3: h_fin WMMA + assemble into w_base =================
// grid (18, 16): n0 = blockIdx.x*256, g = blockIdx.y. 256 threads (8 waves).
// dyn LDS: al[64][128] bf16 (16KB) + bl[256][128] bf16 (64KB) = 80KB
__global__ void hfin_kernel(const bf16_t* __restrict__ hin,
                            const bf16_t* __restrict__ w1bf,
                            const float* __restrict__ hb1,
                            float* __restrict__ wbase) {
  extern __shared__ char smem[];
  bf16_t* al = (bf16_t*)smem;                   // [64][128]
  bf16_t* bl = (bf16_t*)(smem + 64 * 128 * 2);  // [256][128] (n-major, k contiguous)
  int g = blockIdx.y;
  int n0 = blockIdx.x * 256;
  int b = g >> 3, lvl = (g >> 2) & 1, m = g & 3;
  int tid = threadIdx.x;

#if HAVE_ASYNC_LDS
  for (int i = tid; i < (HO * ZD) / 8; i += 256)
    async_copy16(hin + g * (HO * ZD) + i * 8, al + i * 8);
#else
  for (int i = tid; i < HO * ZD; i += 256) al[i] = hin[g * (HO * ZD) + i];
#endif
  for (int i = tid; i < ZD * 256; i += 256) {
    int c = i & 255;
    int k = i >> 8;
    bl[c * ZD + k] = w1bf[k * NFIN + n0 + c];
  }
#if HAVE_ASYNC_LDS
  __builtin_amdgcn_s_wait_asynccnt(0);
#endif
  __syncthreads();

  int wv = tid >> 5, lane = tid & 31, hlf = lane >> 4, mr = lane & 15;
  int mtile = wv >> 1;
  int ntb = (wv & 1) * 8;

  v8f acc[8];
  #pragma unroll
  for (int nt = 0; nt < 8; ++nt) acc[nt] = v8f_zero();

  #pragma unroll
  for (int kt = 0; kt < 4; ++kt) {
    const v8bf* ar = (const v8bf*)(al + (mtile * 16 + mr) * ZD + kt * 32);
    v16bf av = cat8(ar[hlf], ar[2 + hlf]);
    #pragma unroll
    for (int nt = 0; nt < 8; ++nt) {
      int n = (ntb + nt) * 16 + mr;
      const v8bf* br = (const v8bf*)(bl + n * ZD + kt * 32);
      v16bf bv = cat8(br[hlf], br[2 + hlf]);
      acc[nt] = __builtin_amdgcn_wmma_f32_16x16x32_bf16(
          false, av, false, bv, (short)0, acc[nt], false, false);
    }
  }

  // scatter-assemble: kernels[m][a][bb][r][fi][fj] -> w_base[lvl][b][co][ci][r][f]
  int row = m >> 1, colm = m & 1;
  int lb = lvl * 2 + b;
  #pragma unroll
  for (int nt = 0; nt < 8; ++nt) {
    int nglob = n0 + (ntb + nt) * 16 + mr;
    int bb = nglob / 72;
    int rem = nglob - bb * 72;
    int r = rem / 9;
    int ff = rem - r * 9;
    int ci = colm * 64 + bb;
    #pragma unroll
    for (int vv = 0; vv < 8; ++vv) {
      int a = mtile * 16 + vv + 8 * hlf;
      int co = row * 64 + a;
      float val = acc[nt][vv] + hb1[nglob];
      wbase[(((lb * CC + co) * CC + ci) * RR + r) * NF + ff] = val;
    }
  }
}

// ================= K0a: x f32 [b][c][h][w] -> xc bf16 [b][w][h][c] =================
// grid (ctile=4, wtile=4, B*H), 256 threads
__global__ void pack_x_kernel(const float* __restrict__ x, bf16_t* __restrict__ xc) {
  __shared__ bf16_t t[32][33];
  int bh = blockIdx.z;
  int b = bh / HIMG, h = bh % HIMG;
  int c0 = blockIdx.x * 32, w0 = blockIdx.y * 32;
  int tx = threadIdx.x & 31, ty = threadIdx.x >> 5;
  #pragma unroll
  for (int r = 0; r < 32; r += 8)
    t[ty + r][tx] = (bf16_t)x[((size_t)(b * CC + c0 + ty + r) * HIMG + h) * WIMG + w0 + tx];
  __syncthreads();
  #pragma unroll
  for (int r = 0; r < 32; r += 8)
    xc[((size_t)(b * WIMG + w0 + ty + r) * HIMG + h) * CC + c0 + tx] = t[tx][ty + r];
}

// ================= K0b: y bf16 [b][c][w][h] -> yc bf16 [b][w][h][c] =================
// grid (ctile=4, htile=8, B*W), 256 threads
__global__ void pack_y_kernel(const bf16_t* __restrict__ yt, bf16_t* __restrict__ yc) {
  __shared__ bf16_t t[32][33];
  int bw = blockIdx.z;
  int b = bw / WIMG, w = bw % WIMG;
  int c0 = blockIdx.x * 32, h0 = blockIdx.y * 32;
  int tx = threadIdx.x & 31, ty = threadIdx.x >> 5;
  #pragma unroll
  for (int r = 0; r < 32; r += 8)
    t[ty + r][tx] = yt[((size_t)(b * CC + c0 + ty + r) * WIMG + w) * HIMG + h0 + tx];
  __syncthreads();
  #pragma unroll
  for (int r = 0; r < 32; r += 8)
    yc[((size_t)(b * WIMG + w) * HIMG + h0 + ty + r) * CC + c0 + tx] = t[tx][ty + r];
}

// ================= K4: LRI conv (the big WMMA kernel) =================
// grid (coh=2, w=128, b=2), 256 threads (8 waves)
// src is channels-last bf16 [b][w][h][c].
// dyn LDS: xs[3][258][40] bf16 (61920B) + wl[9][64][40] bf16 (46080B) = 108000B
#define XS_STRIDE 40
#define XS_BYTES  (3 * 258 * XS_STRIDE * 2)
__global__ void lri_conv_kernel(const bf16_t* __restrict__ src,   // [b][w][h][ci] bf16
                                const float* __restrict__ wbase,  // [b][co][ci][r][9] f32
                                const float* __restrict__ bng,
                                const float* __restrict__ bnb,
                                bf16_t* __restrict__ dst_t,       // [b][co][w][h] bf16 (conv1)
                                float* __restrict__ dst_f,        // [b][co][h][w] f32  (conv2)
                                int transposed_out) {
  extern __shared__ char smem[];
  bf16_t* xs = (bf16_t*)smem;               // [3][258][40]
  bf16_t* wl = (bf16_t*)(smem + XS_BYTES);  // [9][64][40]

  int coh = blockIdx.x;
  int w = blockIdx.y;
  int b = blockIdx.z;
  int tid = threadIdx.x;

  // radius interpolation params (align_corners=False)
  float srcr = ((float)w + 0.5f) * (8.0f / 128.0f) - 0.5f;
  srcr = fminf(fmaxf(srcr, 0.0f), 7.0f);
  int i0 = (int)floorf(srcr);
  int i1 = min(i0 + 1, RR - 1);
  float t = srcr - (float)i0;

  int wv = tid >> 5, lane = tid & 31, hlf = lane >> 4, mr = lane & 15;
  int mtile = wv & 3;
  int nhb = (wv >> 2) * 8;

  v8f acc[8];
  #pragma unroll
  for (int nt = 0; nt < 8; ++nt) acc[nt] = v8f_zero();

  for (int cc = 0; cc < 4; ++cc) {
    int ci0 = cc * 32;
    __syncthreads();

    // ---- stage activations: 3 columns, circular-H pad, zero-W pad ----
#if HAVE_ASYNC_LDS
    // one 16B lane-copy per quarter-row; both sides 16B aligned
    for (int idx = tid; idx < 3 * 258 * 4; idx += 256) {
      int q = idx & 3;
      int rw = idx >> 2;
      int hrow = rw % 258;
      int dj = rw / 258;
      int wsrc = w + dj - 1;
      bf16_t* ldst = xs + (dj * 258 + hrow) * XS_STRIDE + q * 8;
      if (wsrc >= 0 && wsrc < WIMG) {
        int h = (hrow == 0) ? (HIMG - 1) : ((hrow == HIMG + 1) ? 0 : hrow - 1);
        const bf16_t* gsrc =
            src + ((size_t)(b * WIMG + wsrc) * HIMG + h) * CC + ci0 + q * 8;
        async_copy16(gsrc, ldst);
      } else {
        *(v8bf*)ldst = v8bf_zero();
      }
    }
#else
    for (int idx = tid; idx < 3 * 258 * 32; idx += 256) {
      int cil = idx & 31;
      int rw = idx >> 5;
      int hrow = rw % 258;
      int dj = rw / 258;
      int wsrc = w + dj - 1;
      bf16_t val = (bf16_t)0.0f;
      if (wsrc >= 0 && wsrc < WIMG) {
        int h = (hrow == 0) ? (HIMG - 1) : ((hrow == HIMG + 1) ? 0 : hrow - 1);
        val = src[((size_t)(b * WIMG + wsrc) * HIMG + h) * CC + ci0 + cil];
      }
      xs[(dj * 258 + hrow) * XS_STRIDE + cil] = val;
    }
#endif

    // ---- stage interpolated weights for this column ----
    for (int idx = tid; idx < NF * 64 * 32; idx += 256) {
      int f = idx % 9;
      int r2 = idx / 9;
      int cil = r2 & 31;
      int col = r2 >> 5;
      int co = coh * 64 + col;
      int ci = ci0 + cil;
      const float* wb = wbase + ((((size_t)b * CC + co) * CC + ci) * RR) * NF + f;
      float v0 = wb[i0 * NF];
      float v1 = wb[i1 * NF];
      wl[(f * 64 + col) * XS_STRIDE + cil] = (bf16_t)(v0 + (v1 - v0) * t);
    }
    if (cc < 3)  // hint next ci-chunk of base weights into cache
      __builtin_prefetch(wbase + ((((size_t)b * CC + coh * 64) * CC + (cc + 1) * 32) * RR) * NF, 0, 0);
#if HAVE_ASYNC_LDS
    __builtin_amdgcn_s_wait_asynccnt(0);
#endif
    __syncthreads();

    // ---- compute: 9 taps x 8 N-tiles of 16x16x32 bf16 WMMA ----
    #pragma unroll
    for (int f = 0; f < 9; ++f) {
      int di = f / 3;
      int dj = f - di * 3;
      const v8bf* ar = (const v8bf*)(wl + (f * 64 + mtile * 16 + mr) * XS_STRIDE);
      v16bf av = cat8(ar[hlf], ar[2 + hlf]);
      #pragma unroll
      for (int nt = 0; nt < 8; ++nt) {
        int hh = (nhb + nt) * 16 + mr + di;  // padded row index
        const v8bf* br = (const v8bf*)(xs + (dj * 258 + hh) * XS_STRIDE);
        v16bf bv = cat8(br[hlf], br[2 + hlf]);
        acc[nt] = __builtin_amdgcn_wmma_f32_16x16x32_bf16(
            false, av, false, bv, (short)0, acc[nt], false, false);
      }
    }
  }

  // ---- epilogue: eval BN + ReLU, write out ----
  const float inv = 0.99999500003749968f;  // 1/sqrt(1+1e-5)
  #pragma unroll
  for (int nt = 0; nt < 8; ++nt) {
    int hq = (nhb + nt) * 16 + mr;
    #pragma unroll
    for (int vv = 0; vv < 8; ++vv) {
      int co = coh * 64 + mtile * 16 + vv + 8 * hlf;
      float sc = bng[co] * inv;
      float val = fmaxf(acc[nt][vv] * sc + bnb[co], 0.0f);
      if (transposed_out)
        dst_t[((size_t)(b * CC + co) * WIMG + w) * HIMG + hq] = (bf16_t)val;
      else
        dst_f[((size_t)(b * CC + co) * HIMG + hq) * WIMG + w] = val;
    }
  }
}

// ================= host launcher =================
extern "C" void kernel_launch(void* const* d_in, const int* in_sizes, int n_in,
                              void* d_out, int out_size, void* d_ws, size_t ws_size,
                              hipStream_t stream) {
  const float* x      = (const float*)d_in[0];
  const float* seidel = (const float*)d_in[1];
  const float* m1_W1  = (const float*)d_in[2];
  const float* m1_b1  = (const float*)d_in[3];
  const float* m1_W2  = (const float*)d_in[4];
  const float* m1_b2  = (const float*)d_in[5];
  const float* m2_W1  = (const float*)d_in[6];
  const float* m2_b1  = (const float*)d_in[7];
  const float* m2_W2  = (const float*)d_in[8];
  const float* m2_b2  = (const float*)d_in[9];
  const float* hyp_w2 = (const float*)d_in[10];
  const float* hyp_b2 = (const float*)d_in[11];
  const float* hyp_w1 = (const float*)d_in[12];
  const float* hyp_b1 = (const float*)d_in[13];
  const float* bn1_g  = (const float*)d_in[14];
  const float* bn1_b  = (const float*)d_in[15];
  const float* bn2_g  = (const float*)d_in[16];
  const float* bn2_b  = (const float*)d_in[17];
  float* out = (float*)d_out;

  char* ws = (char*)d_ws;
  float*  emb   = (float*)(ws + 0);          //  16*128 f32        =  8 KB
  bf16_t* hin   = (bf16_t*)(ws + 8192);      //  16*8192 bf16      = 256 KB
  bf16_t* w1bf  = (bf16_t*)(ws + 270336);    //  128*4608 bf16     = 1.125 MB
  float*  wbase = (float*)(ws + 1449984);    //  2*2*128*128*72 f32= 18 MB
  bf16_t* xc    = (bf16_t*)(ws + 20324352);  //  2*128*256*128 bf16= 16 MB  [b][w][h][c]
  bf16_t* y1t   = (bf16_t*)(ws + 37101568);  //  16 MB  [b][c][w][h]
  bf16_t* y1c   = (bf16_t*)(ws + 53878784);  //  16 MB  [b][w][h][c]

  mlp_kernel<<<16, 128, 0, stream>>>(seidel, m1_W1, m1_b1, m1_W2, m1_b2,
                                     m2_W1, m2_b1, m2_W2, m2_b2, emb);
  hin_kernel<<<dim3(32, 16), 256, 0, stream>>>(emb, hyp_w2, hyp_b2, hin);
  cvt_kernel<<<(ZD * NFIN + 255) / 256, 256, 0, stream>>>(hyp_w1, w1bf, ZD * NFIN);
  hfin_kernel<<<dim3(18, 16), 256, 81920, stream>>>(hin, w1bf, hyp_b1, wbase);
  pack_x_kernel<<<dim3(4, 4, BNUM * HIMG), 256, 0, stream>>>(x, xc);

  const size_t lvl_stride = (size_t)BNUM * CC * CC * RR * NF;  // floats per level
  const int conv_lds = XS_BYTES + NF * 64 * XS_STRIDE * 2;     // 108000 B
  lri_conv_kernel<<<dim3(2, WIMG, BNUM), 256, conv_lds, stream>>>(
      xc, wbase, bn1_g, bn1_b, y1t, nullptr, 1);
  pack_y_kernel<<<dim3(4, 8, BNUM * WIMG), 256, 0, stream>>>(y1t, y1c);
  lri_conv_kernel<<<dim3(2, WIMG, BNUM), 256, conv_lds, stream>>>(
      y1c, wbase + lvl_stride, bn2_g, bn2_b, nullptr, out, 0);
}
